// GNN_15968688407024
// MI455X (gfx1250) — matrix operand order repaired and checked by reading, not compiled, not run on hardware
//
#include <hip/hip_runtime.h>

typedef __attribute__((ext_vector_type(16))) _Float16 v16h;
typedef __attribute__((ext_vector_type(8)))  float    v8f;
typedef __attribute__((ext_vector_type(4)))  unsigned int u32x4;
typedef __attribute__((ext_vector_type(8)))  int i32x8;
typedef __attribute__((ext_vector_type(4)))  int i32x4;

// ---------------- degree / normalization ----------------

__global__ void k_deg(const int* __restrict__ dst, float* __restrict__ deg, int E) {
  int e = blockIdx.x * blockDim.x + threadIdx.x;
  if (e < E) atomicAdd(&deg[dst[e]], 1.0f);
}

__global__ void k_dis(float* __restrict__ degdis, int N) {
  int i = blockIdx.x * blockDim.x + threadIdx.x;
  if (i < N) {
    float d = degdis[i] + 1.0f;        // +1 self loop; always > 0
    degdis[i] = rsqrtf(d);
  }
}

// ---------------- split fp32 W into f16 hi/lo, pre-swizzled into WMMA B-frag layout --------
// Element (K, c): n = c>>4, m = c&15, k = K>>5, r = K&31, lane = (r>>4)*16 + m, ii = r&15.
// Stored at frag-major index (n*4+k)*512 + lane*16 + ii  -> per-lane frag is 16 contiguous
// halves (32 B), loadable as a single v16h (2x ds_load_b128) in the GEMM.

__global__ void k_wsplit(const float* __restrict__ W,
                         _Float16* __restrict__ Whi, _Float16* __restrict__ Wlo) {
  int i = blockIdx.x * blockDim.x + threadIdx.x;   // 16384 total
  int K = i >> 7, c = i & 127;
  int n = c >> 4, m = c & 15;
  int k = K >> 5, r = K & 31;
  int lane = ((r >> 4) << 4) + m;
  int ii = r & 15;
  int sidx = (n * 4 + k) * 512 + lane * 16 + ii;
  float w = W[i];
  _Float16 h = (_Float16)w;
  Whi[sidx] = h;
  Wlo[sidx] = (_Float16)(w - (float)h);
}

// ---------------- WMMA GEMM: H[N,128] = X[N,128] @ W[128,128] ----------------
// Error-compensated: X = Xhi+Xlo, W = Whi+Wlo, acc += Xhi*Whi + Xlo*Whi + Xhi*Wlo.
// Block = 256 threads = 8 waves; wave w owns rows [blk*128 + w*16, +16).
// Wsplit points at the CONTIGUOUS 64 KB (Whi || Wlo) swizzled image; staged to LDS by TDM.

__global__ __launch_bounds__(256) void k_gemm(const float* __restrict__ X,
                                              const _Float16* __restrict__ Wsplit,
                                              float* __restrict__ H, int N) {
  __shared__ _Float16 sW[2 * 16384];   // [0..16383]=Whi frags, [16384..]=Wlo frags (64 KB)
  int tid = threadIdx.x;

#if __has_builtin(__builtin_amdgcn_tensor_load_to_lds) && __has_builtin(__builtin_amdgcn_s_wait_tensorcnt)
  // Tensor Data Mover: one wave issues a 16384-dword DMA of (Whi||Wlo) into LDS.
  if (tid < 32) {
    unsigned long long ga = (unsigned long long)(const void*)Wsplit;
    unsigned lds_off = (unsigned)(unsigned long long)(void*)sW;  // flat LDS addr truncates to offset
    u32x4 g0;
    g0.x = 1u;                                              // count=1, user desc
    g0.y = lds_off;                                         // lds_addr
    g0.z = (unsigned)(ga & 0xFFFFFFFFull);                  // global_addr[31:0]
    g0.w = (unsigned)((ga >> 32) & 0x1FFFFFFull) | (2u << 30);  // addr[56:32] | type=2
    const int ndw = 16384;                                  // dwords to move
    i32x8 g1;
    g1[0] = (2 << 16);                                      // wg_mask=0, data_size=4B
    g1[1] = (ndw & 0xFFFF) << 16;                           // tensor_dim0[15:0]
    g1[2] = ((ndw >> 16) & 0xFFFF) | (1 << 16);             // tensor_dim0[31:16] | tensor_dim1=1
    g1[3] = (ndw & 0xFFFF) << 16;                           // tile_dim0 = ndw
    g1[4] = 1;                                              // tile_dim1=1, tile_dim2=0
    g1[5] = ndw;                                            // tensor_dim0_stride[31:0]
    g1[6] = 0;
    g1[7] = 0;
    i32x4 z4 = {0, 0, 0, 0};
    i32x8 z8 = {0, 0, 0, 0, 0, 0, 0, 0};
    __builtin_amdgcn_tensor_load_to_lds(g0, g1, z4, z4, z8, 0);
    __builtin_amdgcn_s_wait_tensorcnt(0);
  }
#else
  {
    const unsigned int* g = (const unsigned int*)Wsplit;
    unsigned int* s = (unsigned int*)sW;
    for (int idx = tid; idx < 16384; idx += 256) s[idx] = g[idx];
  }
#endif
  __syncthreads();

  int wave = tid >> 5;
  int lane = tid & 31;
  int hs   = (lane >> 4) & 1;   // lane-half select
  int m    = lane & 15;
  int row0 = blockIdx.x * 128 + wave * 16;
  if (row0 >= N) return;        // wave-uniform: EXEC stays all-ones for WMMA

  v8f acc[8];
#pragma unroll
  for (int n = 0; n < 8; ++n) acc[n] = (v8f){0.f,0.f,0.f,0.f,0.f,0.f,0.f,0.f};

  const float* xr = X + (size_t)(row0 + m) * 128;
  const _Float16* sWlane = sW + lane * 16;   // lane's slot within each 512-half frag

#pragma unroll
  for (int k = 0; k < 4; ++k) {                    // K tiles of 32
    // A fragment (16x32 f16): lane half hs picks K sub-block {16*hs..+15}
    int kb = 32 * k + 16 * hs;
    float4 f0 = *(const float4*)(xr + kb);
    float4 f1 = *(const float4*)(xr + kb + 4);
    float4 f2 = *(const float4*)(xr + kb + 8);
    float4 f3 = *(const float4*)(xr + kb + 12);
    float av[16] = { f0.x,f0.y,f0.z,f0.w, f1.x,f1.y,f1.z,f1.w,
                     f2.x,f2.y,f2.z,f2.w, f3.x,f3.y,f3.z,f3.w };
    v16h ahi, alo;
#pragma unroll
    for (int i = 0; i < 16; ++i) {
      _Float16 h = (_Float16)av[i];
      ahi[i] = h;
      alo[i] = (_Float16)(av[i] - (float)h);
    }

#pragma unroll
    for (int n = 0; n < 8; ++n) {                  // 8 N-tiles of 16
      int fb = (n * 4 + k) * 512;
      v16h bhi = *(const v16h*)(sWlane + fb);            // 2x ds_load_b128
      v16h blo = *(const v16h*)(sWlane + 16384 + fb);    // 2x ds_load_b128
      acc[n] = __builtin_amdgcn_wmma_f32_16x16x32_f16(false, ahi, false, bhi,
                                                      (short)0, acc[n], false, false);
      acc[n] = __builtin_amdgcn_wmma_f32_16x16x32_f16(false, alo, false, bhi,
                                                      (short)0, acc[n], false, false);
      acc[n] = __builtin_amdgcn_wmma_f32_16x16x32_f16(false, ahi, false, blo,
                                                      (short)0, acc[n], false, false);
    }
  }

  // store C/D: VGPR r -> M = r + 8*hs, N = n*16 + (lane&15)
#pragma unroll
  for (int n = 0; n < 8; ++n) {
    int col = n * 16 + m;
#pragma unroll
    for (int r = 0; r < 8; ++r) {
      int orow = row0 + r + 8 * hs;
      H[(size_t)orow * 128 + col] = acc[n][r];
    }
  }
}

// ---------------- edge scatter: AGG[dst] += H[src] * dis[src]*dis[dst] ----------------
// 32 threads per edge, float4 per thread.

__global__ void k_scatter(const int* __restrict__ src, const int* __restrict__ dst,
                          const float* __restrict__ dis,
                          const float* __restrict__ H, float* __restrict__ AGG, int E) {
  int gid = blockIdx.x * blockDim.x + threadIdx.x;
  int e = gid >> 5;
  if (e >= E) return;
  int c4 = (gid & 31) << 2;
  int s = src[e], d = dst[e];
  float coef = dis[s] * dis[d];
  float4 v = *(const float4*)(H + (size_t)s * 128 + c4);
  float* o = AGG + (size_t)d * 128 + c4;
  atomicAdd(o + 0, v.x * coef);
  atomicAdd(o + 1, v.y * coef);
  atomicAdd(o + 2, v.z * coef);
  atomicAdd(o + 3, v.w * coef);
}

// ---------------- fuse self-loop + bias (+ReLU) ----------------

__global__ void k_finalize(const float* __restrict__ AGG, const float* __restrict__ Hf,
                           const float* __restrict__ dis, const float* __restrict__ bias,
                           float* __restrict__ OUT, int N, int relu) {
  int gid = blockIdx.x * blockDim.x + threadIdx.x;
  if (gid >= N * 128) return;
  int i = gid >> 7, c = gid & 127;
  float di = dis[i];
  float v = AGG[gid] + di * di * Hf[gid] + bias[c];
  if (relu) v = fmaxf(v, 0.0f);
  OUT[gid] = v;
}

// ---------------- column-wise mean reduction ----------------

__global__ void k_reduce(const float* __restrict__ C, float* __restrict__ acc, int N) {
  __shared__ float s[256];
  int t = threadIdx.x;
  int c = t & 127;
  int half = t >> 7;
  int base = blockIdx.x * 64;
  float sum = 0.0f;
  for (int r = half; r < 64; r += 2) {
    int row = base + r;
    if (row < N) sum += C[(size_t)row * 128 + c];
  }
  s[t] = sum;
  __syncthreads();
  if (t < 128) atomicAdd(&acc[t], s[t] + s[t + 128]);
}

__global__ void k_writeout(const float* __restrict__ acc, float* __restrict__ out, int N) {
  int c = threadIdx.x;
  out[c] = acc[c] / (float)N;
}

// ---------------- host side ----------------

extern "C" void kernel_launch(void* const* d_in, const int* in_sizes, int n_in,
                              void* d_out, int out_size, void* d_ws, size_t ws_size,
                              hipStream_t stream) {
  const float* x  = (const float*)d_in[0];
  const int*   ei = (const int*)d_in[1];
  const float* W1 = (const float*)d_in[2];
  const float* b1 = (const float*)d_in[3];
  const float* W2 = (const float*)d_in[4];
  const float* b2 = (const float*)d_in[5];
  int N = in_sizes[0] / 128;
  int E = in_sizes[1] / 2;
  const int* srcI = ei;        // edge_index[0]
  const int* dstI = ei + E;    // edge_index[1]

  char* ws = (char*)d_ws;
  size_t off = 0;
  auto carve = [&](size_t bytes) { size_t p = off; off += (bytes + 255) & ~(size_t)255; return p; };
  float*    dis  = (float*)   (ws + carve((size_t)N * 4));
  _Float16* Whi  = (_Float16*)(ws + carve(16384 * 2));   // contiguous with Wlo: 64 KB image
  _Float16* Wlo  = (_Float16*)(ws + carve(16384 * 2));
  float*    acc  = (float*)   (ws + carve(128 * 4));
  float*    bufA = (float*)   (ws + carve((size_t)N * 128 * 4));  // h = x@W
  float*    bufB = (float*)   (ws + carve((size_t)N * 128 * 4));  // aggregation
  float*    bufC = (float*)   (ws + carve((size_t)N * 128 * 4));  // layer output
  (void)Wlo;

  size_t featBytes = (size_t)N * 128 * 4;
  int nfBlocks = (int)(((size_t)N * 128 + 255) / 256);
  int scBlocks = (int)(((size_t)E * 32 + 255) / 256);

  (void)hipMemsetAsync(dis, 0, (size_t)N * 4, stream);
  (void)hipMemsetAsync(bufB, 0, featBytes, stream);
  (void)hipMemsetAsync(acc, 0, 128 * 4, stream);

  k_deg<<<(E + 255) / 256, 256, 0, stream>>>(dstI, dis, E);
  k_dis<<<(N + 255) / 256, 256, 0, stream>>>(dis, N);

  // ---- layer 1 ----
  k_wsplit<<<64, 256, 0, stream>>>(W1, Whi, Wlo);
  k_gemm<<<(N + 127) / 128, 256, 0, stream>>>(x, Whi, bufA, N);
  k_scatter<<<scBlocks, 256, 0, stream>>>(srcI, dstI, dis, bufA, bufB, E);
  k_finalize<<<nfBlocks, 256, 0, stream>>>(bufB, bufA, dis, b1, bufC, N, 1);

  // ---- layer 2 ----
  (void)hipMemsetAsync(bufB, 0, featBytes, stream);
  k_wsplit<<<64, 256, 0, stream>>>(W2, Whi, Wlo);
  k_gemm<<<(N + 127) / 128, 256, 0, stream>>>(bufC, Whi, bufA, N);
  k_scatter<<<scBlocks, 256, 0, stream>>>(srcI, dstI, dis, bufA, bufB, E);
  k_finalize<<<nfBlocks, 256, 0, stream>>>(bufB, bufA, dis, b2, bufC, N, 0);

  // ---- mean over nodes ----
  k_reduce<<<(N + 63) / 64, 256, 0, stream>>>(bufC, acc, N);
  k_writeout<<<1, 128, 0, stream>>>(acc, (float*)d_out, N);
}
